// CoreFinder_19490561590141
// MI455X (gfx1250) — matrix operand order repaired and checked by reading, not compiled, not run on hardware
//
#include <hip/hip_runtime.h>
#include <hip/hip_bf16.h>
#include <math.h>

// ---------------------------------------------------------------------------
// Problem constants (from the reference)
// ---------------------------------------------------------------------------
#define NV     16000          // N_VARS
#define NC     67200          // N_CLAUSES
#define NNZE   (NC * 3)       // nnz edges (CLAUSE_K = 3)
#define NG     16             // N_GRAPHS
#define LEAK   0.2f
#define EPSV   1e-6f

typedef __bf16 bf16_t;
typedef __attribute__((ext_vector_type(16))) __bf16 v16bf;
typedef __attribute__((ext_vector_type(8)))  float  v8f;

static inline int cdiv(int a, int b) { return (a + b - 1) / b; }

// ---------------------------------------------------------------------------
// Device helpers
// ---------------------------------------------------------------------------
__device__ __forceinline__ float softplusf(float x) {
  return (x > 20.f) ? x : log1pf(expf(x));
}
__device__ __forceinline__ float sigmoidf(float x) {
  return 1.f / (1.f + expf(-x));
}
__device__ __forceinline__ unsigned pcg_hash(unsigned x) {
  x = x * 747796405u + 2891336453u;
  unsigned w = ((x >> ((x >> 28) + 4u)) ^ x) * 277803737u;
  return (w >> 22) ^ w;
}
__device__ __forceinline__ float rng_normal(unsigned seed) {
  unsigned a = pcg_hash(seed);
  unsigned b = pcg_hash(seed ^ 0x9E3779B9u);
  float u1 = fmaxf((a >> 8) * (1.0f / 16777216.0f), 1e-7f);
  float u2 = (b >> 8) * (1.0f / 16777216.0f);
  return sqrtf(-2.f * logf(u1)) * __cosf(6.28318530718f * u2);
}

// ---------------------------------------------------------------------------
// WMMA GEMM v2:  Y[M,N] = act( X[M,K] * Wt^T + bias ),  Wt is [N][K] bf16.
// K multiple of 32, N multiple of 64.
// Block = 256 threads (8 waves), tile 256 rows x 64 cols.
// B tile (64 cols x K bf16, <=32KB) staged once per block into LDS and shared
// by all 8 waves (ds_load_b128 fragments).  Each wave owns 32 rows x 64 cols:
// two A fragments reuse every B fragment -> 8 WMMA per K-step vs 8 global +
// 8 LDS b128 loads.
// A-frag, contiguous K like B: lanes 0-15 hold K 0..15 of row M=lane,
// lanes 16-31 hold K 16..31.  B-frag: lane&15 = column, 16 contiguous
// K per lane, low/high K half by lane>=16.  C/D: row = vgpr + 8*(lane>=16).
// ---------------------------------------------------------------------------
__device__ __forceinline__ v16bf load_a_frag(const float* xrow, int kb1, int kb2) {
  const float4 a0 = *(const float4*)(xrow + kb1);
  const float4 a1 = *(const float4*)(xrow + kb1 + 4);
  const float4 a2 = *(const float4*)(xrow + kb2);
  const float4 a3 = *(const float4*)(xrow + kb2 + 4);
  v16bf a;
  a[0]  = (bf16_t)a0.x; a[1]  = (bf16_t)a0.y; a[2]  = (bf16_t)a0.z; a[3]  = (bf16_t)a0.w;
  a[4]  = (bf16_t)a1.x; a[5]  = (bf16_t)a1.y; a[6]  = (bf16_t)a1.z; a[7]  = (bf16_t)a1.w;
  a[8]  = (bf16_t)a2.x; a[9]  = (bf16_t)a2.y; a[10] = (bf16_t)a2.z; a[11] = (bf16_t)a2.w;
  a[12] = (bf16_t)a3.x; a[13] = (bf16_t)a3.y; a[14] = (bf16_t)a3.z; a[15] = (bf16_t)a3.w;
  return a;
}

template <int ACT>
__global__ void __launch_bounds__(256)
gemm_wmma_kernel(const float* __restrict__ X, const bf16_t* __restrict__ Wt,
                 const float* __restrict__ bias, float* __restrict__ Y,
                 int M, int K, int N) {
  extern __shared__ char smem_raw[];
  bf16_t* sw = (bf16_t*)smem_raw;                 // [64][K] bf16
  const int tid  = threadIdx.x;
  const int wave = tid >> 5;
  const int lane = tid & 31;
  const int half = lane >> 4;
  const int lr   = lane & 15;
  const int col0 = blockIdx.y * 64;

  // ---- cooperative stage of the 64xK weight tile into LDS (contiguous) ----
  {
    const float4* src = (const float4*)(Wt + (size_t)col0 * K);   // 16B = 8 bf16
    float4* dst = (float4*)sw;
    const int chunks = (64 * K) >> 3;
    for (int i = tid; i < chunks; i += 256) dst[i] = src[i];
  }
  __syncthreads();

  // ---- per-wave 32x64 tile ----
  const int row0 = blockIdx.x * 256 + wave * 32;  // no early exit: keep EXEC full
  int ar0 = row0 + lr;       if (ar0 >= M) ar0 = M - 1;
  int ar1 = row0 + 16 + lr;  if (ar1 >= M) ar1 = M - 1;
  const float* xr0 = X + (size_t)ar0 * K;
  const float* xr1 = X + (size_t)ar1 * K;
  const bf16_t* sb = sw + (size_t)lr * K + half * 16;

  v8f acc[8] = {};
  for (int k0 = 0; k0 < K; k0 += 32) {
    __builtin_prefetch(xr0 + k0 + 32, 0, 0);      // global_prefetch next A slice
    __builtin_prefetch(xr1 + k0 + 32, 0, 0);
    const int kb1 = k0 + (half ? 16 : 0);
    const int kb2 = k0 + (half ? 24 : 8);
    v16bf a0 = load_a_frag(xr0, kb1, kb2);
    v16bf a1 = load_a_frag(xr1, kb1, kb2);
#pragma unroll
    for (int s = 0; s < 4; ++s) {
      v16bf b = *(const v16bf*)(sb + (size_t)(s * 16) * K + k0);  // 2x ds_load_b128
      acc[s]     = __builtin_amdgcn_wmma_f32_16x16x32_bf16(false, a0, false, b, (short)0, acc[s],     false, false);
      acc[4 + s] = __builtin_amdgcn_wmma_f32_16x16x32_bf16(false, a1, false, b, (short)0, acc[4 + s], false, false);
    }
  }
  // ---- epilogue: C/D layout row = vgpr + 8*half ----
#pragma unroll
  for (int h = 0; h < 2; ++h) {
#pragma unroll
    for (int v = 0; v < 8; ++v) {
      const int orow = row0 + h * 16 + v + half * 8;
      if (orow >= M) continue;
#pragma unroll
      for (int s = 0; s < 4; ++s) {
        const int col = col0 + s * 16 + lr;
        float val = acc[h * 4 + s][v] + bias[col];
        if (ACT == 1) val = (val > 0.f) ? val : LEAK * val;
        Y[(size_t)orow * N + col] = val;
      }
    }
  }
}

// W[K,N] f32 -> Wt[N,Kp] bf16 (zero padded rows K..Kp)
__global__ void pack_wt_kernel(const float* __restrict__ W, bf16_t* __restrict__ Wt,
                               int K, int N, int Kp) {
  int idx = blockIdx.x * blockDim.x + threadIdx.x;
  if (idx >= N * Kp) return;
  int n = idx / Kp, k = idx % Kp;
  Wt[(size_t)n * Kp + k] = (k < K) ? (bf16_t)W[(size_t)k * N + n] : (bf16_t)0.f;
}

// out[r] = dot(H[r,:64], w) + b   (final 64 -> 1 layers)
__global__ void rowdot_kernel(const float* __restrict__ H, const float* __restrict__ w,
                              const float* __restrict__ b, float* __restrict__ out, int R) {
  int r = blockIdx.x * blockDim.x + threadIdx.x;
  if (r >= R) return;
  const float* h = H + (size_t)r * 64;
  float acc = b[0];
#pragma unroll
  for (int f = 0; f < 64; ++f) acc = fmaf(h[f], w[f], acc);
  out[r] = acc;
}

// ---------------------------------------------------------------------------
// Structural / elementwise kernels
// ---------------------------------------------------------------------------
__global__ void fill_kernel(float* p, float v, int n) {
  int i = blockIdx.x * blockDim.x + threadIdx.x;
  if (i < n) p[i] = v;
}
__global__ void copy_kernel(const float* s, float* d, int n) {
  int i = blockIdx.x * blockDim.x + threadIdx.x;
  if (i < n) d[i] = s[i];
}
__global__ void scale_kernel(float* p, float s, int n) {
  int i = blockIdx.x * blockDim.x + threadIdx.x;
  if (i < n) p[i] *= s;
}
__global__ void deg_scatter_kernel(const int* __restrict__ lit, float* deg) {
  int e = blockIdx.x * blockDim.x + threadIdx.x;
  if (e < NNZE) atomicAdd(&deg[lit[e]], 1.f);
}
__global__ void degw_kernel(const float* __restrict__ deg, float* dw, float* vdw) {
  int l = blockIdx.x * blockDim.x + threadIdx.x;
  if (l >= 2 * NV) return;
  dw[l] = rsqrtf(fmaxf(deg[l], 1.f));
  if (l < NV) vdw[l] = 4.f * rsqrtf(fmaxf(deg[l] + deg[l + NV], 1.f));
}
__global__ void gcount_kernel(const int* __restrict__ gid, float* cnt, int R) {
  int r = blockIdx.x * blockDim.x + threadIdx.x;
  if (r < R) atomicAdd(&cnt[gid[r]], 1.f);
}
__global__ void inv_kernel(const float* cnt, float* inv) {
  int g = threadIdx.x;
  if (g < NG) inv[g] = 1.f / fmaxf(cnt[g], 1.f);
}

// v1 = [variables | noise(4) | zero pad] -> NV x 96
__global__ void v1_kernel(const float* __restrict__ vars, float* v1, unsigned seed) {
  int i = blockIdx.x * blockDim.x + threadIdx.x;
  if (i >= NV * 96) return;
  int r = i / 96, c = i % 96;
  float v;
  if (c < 64)       v = vars[(size_t)r * 64 + c];
  else if (c < 68)  v = rng_normal(seed + (unsigned)(r * 4 + (c - 64)));
  else              v = 0.f;
  v1[i] = v;
}

// per-clause: s = sum softplus(+-q), cl = exp(-s); build clause_unit rows
__global__ void clause_unit_kernel(const float* __restrict__ query, const int* __restrict__ lit,
                                   const float* __restrict__ cstate, float* __restrict__ cl,
                                   float* __restrict__ cunit) {
  int c = blockIdx.x, f = threadIdx.x;
  float s = 0.f;
#pragma unroll
  for (int j = 0; j < 3; ++j) {
    int l = lit[3 * c + j];
    int v = (l < NV) ? l : (l - NV);
    float sg = (l < NV) ? 1.f : -1.f;
    s += softplusf(sg * query[(size_t)v * 64 + f]);
  }
  float clv = expf(-s);
  cl[(size_t)c * 64 + f] = clv;
  cunit[(size_t)c * 128 + f]      = cstate[(size_t)c * 64 + f];
  cunit[(size_t)c * 128 + 64 + f] = 4.f * clv;
}

// grad_q[v] += (l<NV ? -cl*sigmoid(q) : +cl*sigmoid(-q))   (201600 edges x 64)
__global__ void grad_scatter_kernel(const int* __restrict__ lit, const int* __restrict__ cidx,
                                    const float* __restrict__ cl, const float* __restrict__ query,
                                    float* grad_q) {
  int t = blockIdx.x * blockDim.x + threadIdx.x;
  if (t >= NNZE * 64) return;
  int e = t >> 6, f = t & 63;
  int l = lit[e], c = cidx[e];
  float clv = cl[(size_t)c * 64 + f];
  if (l < NV) {
    float q = query[(size_t)l * 64 + f];
    atomicAdd(&grad_q[(size_t)l * 64 + f], -clv * sigmoidf(q));
  } else {
    int v = l - NV;
    float q = query[(size_t)v * 64 + f];
    atomicAdd(&grad_q[(size_t)v * 64 + f], clv * sigmoidf(-q));
  }
}

// varloss[lit] += clause_data[c, 0:64]
__global__ void varloss_scatter_kernel(const int* __restrict__ lit, const int* __restrict__ cidx,
                                       const float* __restrict__ cdata, float* varloss) {
  int t = blockIdx.x * blockDim.x + threadIdx.x;
  if (t >= NNZE * 64) return;
  int e = t >> 6, f = t & 63;
  atomicAdd(&varloss[(size_t)lit[e] * 64 + f], cdata[(size_t)cidx[e] * 128 + f]);
}

// unit = [grad*vdw | variables | vl_pos*dw | vl_neg*dw]  -> NV x 256
__global__ void unit_kernel(const float* __restrict__ grad_q, const float* __restrict__ vdw,
                            const float* __restrict__ vars, const float* __restrict__ varloss,
                            const float* __restrict__ dw, float* unit) {
  int t = blockIdx.x * blockDim.x + threadIdx.x;
  if (t >= NV * 64) return;
  int v = t >> 6, f = t & 63;
  float* u = unit + (size_t)v * 256;
  u[f]       = grad_q[(size_t)v * 64 + f] * vdw[v];
  u[64 + f]  = vars[(size_t)v * 64 + f];
  u[128 + f] = varloss[(size_t)v * 64 + f] * dw[v];
  u[192 + f] = varloss[(size_t)(v + NV) * 64 + f] * dw[NV + v];
}

// ---- pair norm (4 phases) --------------------------------------------------
__global__ void pn_sum_kernel(const float* __restrict__ X, int ld, int off,
                              const int* __restrict__ gid, float* gsum, int R) {
  int t = blockIdx.x * blockDim.x + threadIdx.x;
  if (t >= R * 64) return;
  int r = t >> 6, f = t & 63;
  atomicAdd(&gsum[gid[r] * 64 + f], X[(size_t)r * ld + off + f]);
}
__global__ void pn_center_kernel(const float* __restrict__ X, int ld, int off,
                                 const int* __restrict__ gid, const float* __restrict__ inv,
                                 const float* __restrict__ gsum, float* xc, float* gsq, int R) {
  int t = blockIdx.x * blockDim.x + threadIdx.x;
  if (t >= R * 64) return;
  int r = t >> 6, f = t & 63, g = gid[r];
  float m = gsum[g * 64 + f] * inv[g];
  float v = X[(size_t)r * ld + off + f] - m;
  xc[(size_t)r * 64 + f] = v;
  atomicAdd(&gsq[g * 64 + f], v * v);
}
__global__ void pn_var_kernel(const float* __restrict__ gsq, const float* __restrict__ inv,
                              float* gvar) {
  int g = threadIdx.x;
  if (g >= NG) return;
  float s = 0.f;
  for (int f = 0; f < 64; ++f) s += gsq[g * 64 + f];
  gvar[g] = s * inv[g] * (1.f / 64.f);
}
// state = 0.25*(gain*xc*rsqrt(var+eps)+bias) + decay*state
__global__ void pn_apply_kernel(const float* __restrict__ xc, const int* __restrict__ gid,
                                const float* __restrict__ gvar, const float* __restrict__ gain,
                                const float* __restrict__ bias, float* state, float decay, int R) {
  int t = blockIdx.x * blockDim.x + threadIdx.x;
  if (t >= R * 64) return;
  int r = t >> 6, f = t & 63;
  float rs = rsqrtf(gvar[gid[r]] + EPSV);
  float y = gain[f] * xc[(size_t)r * 64 + f] * rs + bias[f];
  state[(size_t)r * 64 + f] = 0.25f * y + decay * state[(size_t)r * 64 + f];
}

// solver mixed loss accumulation (per clause, scalar logits)
__global__ void solver_loss_kernel(const int* __restrict__ lit, const float* __restrict__ logit,
                                   float* loss_acc) {
  int c = blockIdx.x * blockDim.x + threadIdx.x;
  if (c >= NC) return;
  float s = 0.f;
#pragma unroll
  for (int j = 0; j < 3; ++j) {
    int l = lit[3 * c + j];
    float q = (l < NV) ? logit[l] : -logit[l - NV];
    s += softplusf(q);
  }
  float cv = expf(-s);
  loss_acc[c] += cv * (-logf(1.f - cv + 1e-10f));
}
__global__ void mask_kernel(const float* __restrict__ logit, float* out, unsigned seed) {
  int c = blockIdx.x * blockDim.x + threadIdx.x;
  if (c >= NC) return;
  out[c] = sigmoidf(logit[c] + rng_normal(seed + (unsigned)c));
}
__global__ void finalize_loss_kernel(const float* __restrict__ loss, float* out) {
  int c = blockIdx.x * blockDim.x + threadIdx.x;
  if (c < NC) out[c] = loss[c] * (1.f / 8.f);
}

// ---------------------------------------------------------------------------
// Host-side orchestration
// ---------------------------------------------------------------------------
struct SubP {
  const float *cmW0, *cmb0, *cmW1, *cmb1;
  const float *cnB, *cnG;
  const float *outW0, *outb0, *outW1, *outb1;
  const float *ugW0, *ugb0, *ugW1, *ugb1, *ugW2, *ugb2;
  const float *vnB, *vnG;
  const float *vqW0, *vqb0, *vqW1, *vqb1;
};

struct WS {
  float *deg, *dw, *vdw, *inv_v, *inv_c, *cntv, *cntc;
  float *vars, *cstate, *v1, *qh, *query;
  float *cl, *cunit, *chid, *cdata;
  float *grad_q, *varloss, *unit, *uh1, *uh2, *newv, *xc_v;
  float *gsum, *gsq, *gvar, *logits, *loss;
  bf16_t *wt_vq0, *wt_vq1, *wt_cm0, *wt_cm1, *wt_ug0, *wt_ug1, *wt_ug2, *wt_out0;
};

static void gemm(hipStream_t st, const float* X, const bf16_t* Wt, const float* b,
                 float* Y, int M, int K, int N, int act) {
  dim3 g(cdiv(M, 256), N / 64), blk(256);
  const size_t smem = (size_t)64 * K * sizeof(bf16_t);
  if (act) gemm_wmma_kernel<1><<<g, blk, smem, st>>>(X, Wt, b, Y, M, K, N);
  else     gemm_wmma_kernel<0><<<g, blk, smem, st>>>(X, Wt, b, Y, M, K, N);
}

static void run_sub(hipStream_t st, const SubP& P, bool isMin,
                    const int* lit, const int* cidx, const int* vgid, const int* cgid,
                    WS& w, float* d_out, unsigned seed0) {
  auto pack = [&](const float* W, bf16_t* Wt, int K, int N, int Kp) {
    pack_wt_kernel<<<cdiv(N * Kp, 256), 256, 0, st>>>(W, Wt, K, N, Kp);
  };
  pack(P.vqW0, w.wt_vq0, 68, 64, 96);
  pack(P.vqW1, w.wt_vq1, 64, 64, 64);
  pack(P.cmW0, w.wt_cm0, 128, 128, 128);
  pack(P.cmW1, w.wt_cm1, 128, 128, 128);
  pack(P.ugW0, w.wt_ug0, 256, 128, 256);
  pack(P.ugW1, w.wt_ug1, 128, 128, 128);
  pack(P.ugW2, w.wt_ug2, 128, 64, 128);
  pack(P.outW0, w.wt_out0, 64, 64, 64);

  fill_kernel<<<cdiv(NV * 64, 256), 256, 0, st>>>(w.vars, 1.f, NV * 64);
  fill_kernel<<<cdiv(NC * 64, 256), 256, 0, st>>>(w.cstate, 1.f, NC * 64);
  if (!isMin) hipMemsetAsync(w.loss, 0, (size_t)NC * 4, st);

  const int EG = cdiv(NNZE * 64, 256);
  for (int r = 0; r < 8; ++r) {
    unsigned seed = seed0 + 0x10001u * (unsigned)r;
    hipMemsetAsync(w.grad_q, 0, (size_t)NV * 64 * 4, st);
    hipMemsetAsync(w.varloss, 0, (size_t)2 * NV * 64 * 4, st);

    v1_kernel<<<cdiv(NV * 96, 256), 256, 0, st>>>(w.vars, w.v1, seed);
    gemm(st, w.v1, w.wt_vq0, P.vqb0, w.qh, NV, 96, 64, 1);
    gemm(st, w.qh, w.wt_vq1, P.vqb1, w.query, NV, 64, 64, 0);

    clause_unit_kernel<<<NC, 64, 0, st>>>(w.query, lit, w.cstate, w.cl, w.cunit);
    grad_scatter_kernel<<<EG, 256, 0, st>>>(lit, cidx, w.cl, w.query, w.grad_q);

    gemm(st, w.cunit, w.wt_cm0, P.cmb0, w.chid, NC, 128, 128, 1);
    gemm(st, w.chid, w.wt_cm1, P.cmb1, w.cdata, NC, 128, 128, 0);

    // clause pair-norm on cdata[:,64:128]; centered tile reuses w.cl
    hipMemsetAsync(w.gsum, 0, NG * 64 * 4, st);
    hipMemsetAsync(w.gsq, 0, NG * 64 * 4, st);
    pn_sum_kernel<<<cdiv(NC * 64, 256), 256, 0, st>>>(w.cdata, 128, 64, cgid, w.gsum, NC);
    pn_center_kernel<<<cdiv(NC * 64, 256), 256, 0, st>>>(w.cdata, 128, 64, cgid, w.inv_c,
                                                         w.gsum, w.cl, w.gsq, NC);
    pn_var_kernel<<<1, NG, 0, st>>>(w.gsq, w.inv_c, w.gvar);
    pn_apply_kernel<<<cdiv(NC * 64, 256), 256, 0, st>>>(w.cl, cgid, w.gvar, P.cnG, P.cnB,
                                                        w.cstate, 0.1f, NC);

    varloss_scatter_kernel<<<EG, 256, 0, st>>>(lit, cidx, w.cdata, w.varloss);
    unit_kernel<<<cdiv(NV * 64, 256), 256, 0, st>>>(w.grad_q, w.vdw, w.vars, w.varloss,
                                                    w.dw, w.unit);
    gemm(st, w.unit, w.wt_ug0, P.ugb0, w.uh1, NV, 256, 128, 1);
    gemm(st, w.uh1, w.wt_ug1, P.ugb1, w.uh2, NV, 128, 128, 1);
    gemm(st, w.uh2, w.wt_ug2, P.ugb2, w.newv, NV, 128, 64, 0);

    hipMemsetAsync(w.gsum, 0, NG * 64 * 4, st);
    hipMemsetAsync(w.gsq, 0, NG * 64 * 4, st);
    pn_sum_kernel<<<cdiv(NV * 64, 256), 256, 0, st>>>(w.newv, 64, 0, vgid, w.gsum, NV);
    pn_center_kernel<<<cdiv(NV * 64, 256), 256, 0, st>>>(w.newv, 64, 0, vgid, w.inv_v,
                                                         w.gsum, w.xc_v, w.gsq, NV);
    pn_var_kernel<<<1, NG, 0, st>>>(w.gsq, w.inv_v, w.gvar);
    pn_apply_kernel<<<cdiv(NV * 64, 256), 256, 0, st>>>(w.xc_v, vgid, w.gvar, P.vnG, P.vnB,
                                                        w.vars, 0.1f, NV);

    if (!isMin) {  // solver: logits over variables, accumulate mixed loss
      gemm(st, w.vars, w.wt_out0, P.outb0, w.chid, NV, 64, 64, 1);
      rowdot_kernel<<<cdiv(NV, 256), 256, 0, st>>>(w.chid, P.outW1, P.outb1, w.logits, NV);
      solver_loss_kernel<<<cdiv(NC, 256), 256, 0, st>>>(lit, w.logits, w.loss);
    } else {       // minimizer: logits over clauses, then clauses *= 0.2
      gemm(st, w.cstate, w.wt_out0, P.outb0, w.chid, NC, 64, 64, 1);
      rowdot_kernel<<<cdiv(NC, 256), 256, 0, st>>>(w.chid, P.outW1, P.outb1, w.logits, NC);
      scale_kernel<<<cdiv(NC * 64, 256), 256, 0, st>>>(w.cstate, 0.2f, NC * 64);
    }
  }
  if (isMin) {
    mask_kernel<<<cdiv(NC, 256), 256, 0, st>>>(w.logits, d_out, 0xC0FFEEu);
  } else {
    copy_kernel<<<cdiv(NV, 256), 256, 0, st>>>(w.logits, d_out + NC, NV);
    finalize_loss_kernel<<<cdiv(NC, 256), 256, 0, st>>>(w.loss, d_out + NC + NV);
  }
}

// ---------------------------------------------------------------------------
// Entry point.  d_in layout (setup_inputs order, params tree-flattened with
// sorted dict keys, lists in order, (W,b) tuples in order):
//   0 lit_idx, 1 clause_idx, 2 var_graph_id, 3 clause_graph_id,
//   4 n_clauses, 5 n_graphs,
//   6.. minimizer leaves (22), 28.. solver leaves (22)
// d_out = [clauses_mask(NC) | last_logits(NV) | unsup_loss(NC)]
// ---------------------------------------------------------------------------
extern "C" void kernel_launch(void* const* d_in, const int* in_sizes, int n_in,
                              void* d_out, int out_size, void* d_ws, size_t ws_size,
                              hipStream_t stream) {
  (void)in_sizes; (void)n_in; (void)out_size; (void)ws_size;
  const int* lit  = (const int*)d_in[0];
  const int* cidx = (const int*)d_in[1];
  const int* vgid = (const int*)d_in[2];
  const int* cgid = (const int*)d_in[3];
  const float* const* L = (const float* const*)(d_in + 6);

  // minimizer: clause_mlp, clauses_norm{bias,gain}, clauses_output, update_gate,
  //            variables_norm{bias,gain}, variables_query
  SubP Pm = { L[0], L[1], L[2], L[3],  L[4], L[5],
              L[6], L[7], L[8], L[9],
              L[10], L[11], L[12], L[13], L[14], L[15],
              L[16], L[17], L[18], L[19], L[20], L[21] };
  // solver: clause_mlp, clauses_norm, update_gate, variables_norm,
  //         variables_output, variables_query
  const float* const* S = L + 22;
  SubP Ps = { S[0], S[1], S[2], S[3],  S[4], S[5],
              S[14], S[15], S[16], S[17],
              S[6], S[7], S[8], S[9], S[10], S[11],
              S[12], S[13], S[18], S[19], S[20], S[21] };

  // workspace bump allocator (256B aligned)
  char* base = (char*)d_ws;
  size_t off = 0;
  auto takef = [&](size_t n) -> float* {
    size_t a = (off + 255) & ~(size_t)255; off = a + n * 4; return (float*)(base + a);
  };
  auto takeb = [&](size_t n) -> bf16_t* {
    size_t a = (off + 255) & ~(size_t)255; off = a + n * 2; return (bf16_t*)(base + a);
  };
  WS w;
  w.deg = takef(2 * NV);   w.dw = takef(2 * NV);   w.vdw = takef(NV);
  w.inv_v = takef(NG);     w.inv_c = takef(NG);    w.cntv = takef(NG);  w.cntc = takef(NG);
  w.vars = takef((size_t)NV * 64);    w.cstate = takef((size_t)NC * 64);
  w.v1 = takef((size_t)NV * 96);      w.qh = takef((size_t)NV * 64);
  w.query = takef((size_t)NV * 64);
  w.cl = takef((size_t)NC * 64);      w.cunit = takef((size_t)NC * 128);
  w.chid = takef((size_t)NC * 128);   w.cdata = takef((size_t)NC * 128);
  w.grad_q = takef((size_t)NV * 64);  w.varloss = takef((size_t)2 * NV * 64);
  w.unit = takef((size_t)NV * 256);   w.uh1 = takef((size_t)NV * 128);
  w.uh2 = takef((size_t)NV * 128);    w.newv = takef((size_t)NV * 64);
  w.xc_v = takef((size_t)NV * 64);
  w.gsum = takef(NG * 64); w.gsq = takef(NG * 64); w.gvar = takef(NG);
  w.logits = takef(NC);    w.loss = takef(NC);
  w.wt_vq0 = takeb(64 * 96);   w.wt_vq1 = takeb(64 * 64);
  w.wt_cm0 = takeb(128 * 128); w.wt_cm1 = takeb(128 * 128);
  w.wt_ug0 = takeb(128 * 256); w.wt_ug1 = takeb(128 * 128);
  w.wt_ug2 = takeb(64 * 128);  w.wt_out0 = takeb(64 * 64);

  // shared precompute: degrees + group inverse counts
  hipMemsetAsync(w.deg, 0, (size_t)2 * NV * 4, stream);
  hipMemsetAsync(w.cntv, 0, NG * 4, stream);
  hipMemsetAsync(w.cntc, 0, NG * 4, stream);
  deg_scatter_kernel<<<cdiv(NNZE, 256), 256, 0, stream>>>(lit, w.deg);
  gcount_kernel<<<cdiv(NV, 256), 256, 0, stream>>>(vgid, w.cntv, NV);
  gcount_kernel<<<cdiv(NC, 256), 256, 0, stream>>>(cgid, w.cntc, NC);
  degw_kernel<<<cdiv(2 * NV, 256), 256, 0, stream>>>(w.deg, w.dw, w.vdw);
  inv_kernel<<<1, NG, 0, stream>>>(w.cntv, w.inv_v);
  inv_kernel<<<1, NG, 0, stream>>>(w.cntc, w.inv_c);

  run_sub(stream, Pm, true,  lit, cidx, vgid, cgid, w, (float*)d_out, 0x13572468u);
  run_sub(stream, Ps, false, lit, cidx, vgid, cgid, w, (float*)d_out, 0xA5A5C3D1u);
}
